// VQ_VAE_57475252355204
// MI455X (gfx1250) — matrix operand order repaired and checked by reading, not compiled, not run on hardware
//
#include <hip/hip_runtime.h>
#include <hip/hip_bf16.h>
#include <cstdint>

typedef __attribute__((ext_vector_type(16))) __bf16 v16bf;
typedef __attribute__((ext_vector_type(8)))  float  v8f;

static __device__ __forceinline__ unsigned short f2bf(float f) {
    union { float f; unsigned int u; } c; c.f = f;
    unsigned int u = c.u;
    u += 0x7FFFu + ((u >> 16) & 1u);   // round-to-nearest-even
    return (unsigned short)(u >> 16);
}

// f32 (rows x cols) -> bf16 (rowsPad x colsPad), zero-filled padding.
__global__ void cast_pad_kernel(const float* __restrict__ src,
                                unsigned short* __restrict__ dst,
                                int rows, int cols, int colsPad, long long total) {
    long long i = (long long)blockIdx.x * blockDim.x + threadIdx.x;
    if (i >= total) return;
    int r = (int)(i / colsPad), c = (int)(i % colsPad);
    unsigned short v = 0;
    if (r < rows && c < cols) v = f2bf(src[(size_t)r * cols + c]);
    dst[i] = v;
}

// emb_w (d,n) 256x256 f32 -> Wt[n][d] bf16 (B-fragment K-reads contiguous)
__global__ void cast_transpose256(const float* __restrict__ W,
                                  unsigned short* __restrict__ Wt) {
    int d = threadIdx.x, n = blockIdx.x;
    Wt[n * 256 + d] = f2bf(W[d * 256 + n]);
}

// wn[k] = sum_d W[d,k]^2 in fp32
__global__ void wnorm_kernel(const float* __restrict__ W, float* __restrict__ wn) {
    int k = threadIdx.x;
    float s = 0.f;
    for (int d = 0; d < 256; ++d) { float v = W[d * 256 + k]; s += v * v; }
    wn[k] = s;
}

// C[m,n] = act(sum_k A[m,k]*W[n,k] + bias[n]).
// A: M x Kp bf16 (zero-padded K), W: (gridDim.y*64) x Kp bf16 (zero-padded rows/K).
// 64x64 block tile, 4 waves x (32x32), K-step 32, double-buffered LDS filled by
// GLOBAL_LOAD_ASYNC_TO_LDS_B128 (ASYNCcnt), per CDNA5 ISA.
template <int ACT, bool WF32, bool WBF16>
__global__ __launch_bounds__(128) void gemm_kernel(
    const unsigned short* __restrict__ A, const unsigned short* __restrict__ W,
    const float* __restrict__ bias, float* __restrict__ outF,
    unsigned short* __restrict__ outB, int N, int NBpad, int Kp) {
    __shared__ __align__(16) unsigned short Alds[2][64 * 48];
    __shared__ __align__(16) unsigned short Wlds[2][64 * 48];
    const int tid = threadIdx.x;
    const int Mbase = blockIdx.x * 64;
    const int Nbase = blockIdx.y * 64;
    const int wave = tid >> 5, lane = tid & 31;
    const int wm = (wave >> 1) * 32, wn = (wave & 1) * 32;
    const int lm = lane & 15, lh = lane >> 4;
    const int nk = Kp >> 5;

    // Issue 4 async direct-to-LDS 16B loads (2 A + 2 W) for K-step `step`.
    auto stage = [&](int buf, int step) {
        int ktk = step * 32;
        #pragma unroll
        for (int it = 0; it < 2; ++it) {
            int slot = tid + it * 128;            // 0..255
            int row = slot >> 2, kq = slot & 3;   // row in tile, 16B chunk
            int k = ktk + kq * 8;
            unsigned aoff = (unsigned)(((unsigned)(Mbase + row) * (unsigned)Kp + k) * 2u);
            unsigned woff = (unsigned)(((unsigned)(Nbase + row) * (unsigned)Kp + k) * 2u);
            unsigned lda = (unsigned)(uintptr_t)&Alds[buf][row * 48 + kq * 8];
            unsigned ldw = (unsigned)(uintptr_t)&Wlds[buf][row * 48 + kq * 8];
            asm volatile("global_load_async_to_lds_b128 %0, %1, %2"
                         :: "v"(lda), "v"(aoff), "s"(A) : "memory");
            asm volatile("global_load_async_to_lds_b128 %0, %1, %2"
                         :: "v"(ldw), "v"(woff), "s"(W) : "memory");
        }
    };

    v8f acc[2][2] = {};
    stage(0, 0);
    int cur = 0;
    for (int s = 0; s < nk; ++s) {
        if (s + 1 < nk) {
            stage(cur ^ 1, s + 1);   // prefetch next tile while this one computes
            asm volatile("s_wait_asynccnt 0x4" ::: "memory");  // cur tile landed
        } else {
            asm volatile("s_wait_asynccnt 0x0" ::: "memory");
        }
        __syncthreads();

        union FR { v16bf v; unsigned int u[8]; };
        FR afr[2], bfr[2];
        #pragma unroll
        for (int t = 0; t < 2; ++t) {
            int arow = wm + t * 16 + lm;
            int nrow = wn + t * 16 + lm;
            #pragma unroll
            for (int j = 0; j < 8; ++j) {
                int ka = 16 * (j >> 2) + 8 * lh + 2 * (j & 3);  // A 16x32 bf16 layout
                afr[t].u[j] = *reinterpret_cast<const unsigned int*>(&Alds[cur][arow * 48 + ka]);
                int kb = 16 * (j >> 2) + 8 * lh + 2 * (j & 3);  // B 32x16 bf16 layout
                bfr[t].u[j] = *reinterpret_cast<const unsigned int*>(&Wlds[cur][nrow * 48 + kb]);
            }
        }
        #pragma unroll
        for (int i = 0; i < 2; ++i)
            #pragma unroll
            for (int j = 0; j < 2; ++j)
                acc[i][j] = __builtin_amdgcn_wmma_f32_16x16x32_bf16(
                    false, afr[i].v, false, bfr[j].v, (short)0, acc[i][j], false, false);
        __syncthreads();
        cur ^= 1;
    }

    #pragma unroll
    for (int j = 0; j < 2; ++j) {
        int col = Nbase + wn + j * 16 + lm;
        float bv = (col < N) ? bias[col] : 0.f;
        #pragma unroll
        for (int i = 0; i < 2; ++i) {
            #pragma unroll
            for (int v = 0; v < 8; ++v) {
                int row = Mbase + wm + i * 16 + 8 * lh + v;
                float x = acc[i][j][v] + bv;
                if (ACT == 1) x = x > 0.f ? x : 0.f;
                else if (ACT == 2) x = 1.f / (1.f + __expf(-x));
                if (WF32 && col < N) outF[(size_t)row * N + col] = x;
                if (WBF16 && col < NBpad)
                    outB[(size_t)row * NBpad + col] = (col < N) ? f2bf(x) : (unsigned short)0;
            }
        }
    }
}

// Nearest-embedding quantize. Rows r = b*8+p (262144), A[r,d]=z[b, d*8+p].
// Scores S = A (262144x256) x W (256x256) via WMMA; argmin_k (wn[k]-2*S[r,k]).
// A fragments VGPR-resident; B fragments double-buffered across n-tiles.
__global__ __launch_bounds__(128) void quantize_kernel(
    const unsigned short* __restrict__ Zbf,  // B x 2048 bf16
    const unsigned short* __restrict__ Wt,   // 256 x 256 bf16, Wt[n][d]
    const float* __restrict__ Wf,            // emb_w f32 (d,n)
    const float* __restrict__ wn,            // 256 fp32 col norms
    float* __restrict__ embOut,              // B x 2048 f32
    unsigned short* __restrict__ Qbf) {      // B x 2048 bf16 (feeds fc3)
    __shared__ __align__(16) unsigned short zl[8 * 2048];  // 8 z rows = 32 KB
    __shared__ int idx_lds[64];
    const int tid = threadIdx.x;
    const int wave = tid >> 5, lane = tid & 31;
    const int lm = lane & 15, lh = lane >> 4;
    const size_t rowBase = (size_t)blockIdx.x * 64;  // 8 consecutive b's

    {   // coalesced stage of 8 contiguous z rows
        const uint4* src = reinterpret_cast<const uint4*>(Zbf + (rowBase >> 3) * 2048);
        uint4* dst = reinterpret_cast<uint4*>(zl);
        for (int i = tid; i < 2048; i += 128) dst[i] = src[i];
    }
    __syncthreads();

    union FR { v16bf v; unsigned int u[8]; };
    FR afr[8];  // this wave's 16 rows x K=256, resident across the n loop
    {
        int lr = wave * 16 + lm;
        int zoff = ((lr >> 3) * 2048) + (lr & 7);
        #pragma unroll
        for (int kt = 0; kt < 8; ++kt) {
            #pragma unroll
            for (int j = 0; j < 8; ++j) {
                int d0 = kt * 32 + 16 * (j >> 2) + 8 * lh + 2 * (j & 3);
                unsigned int lo = zl[zoff + 8 * d0];
                unsigned int hi = zl[zoff + 8 * (d0 + 1)];
                afr[kt].u[j] = lo | (hi << 16);
            }
        }
    }

    auto loadB = [&](int nt, FR* bf) {
        int nrow = nt * 16 + lm;
        #pragma unroll
        for (int kt = 0; kt < 8; ++kt)
            #pragma unroll
            for (int j = 0; j < 8; ++j) {
                int kb = kt * 32 + 16 * (j >> 2) + 8 * lh + 2 * (j & 3);
                bf[kt].u[j] = *reinterpret_cast<const unsigned int*>(&Wt[nrow * 256 + kb]);
            }
    };

    float bestv[8]; int besti[8];
    #pragma unroll
    for (int v = 0; v < 8; ++v) { bestv[v] = 3.0e38f; besti[v] = 0; }

    FR bfr[2][8];
    loadB(0, bfr[0]);
    #pragma unroll
    for (int nt = 0; nt < 16; ++nt) {
        if (nt + 1 < 16) loadB(nt + 1, bfr[(nt + 1) & 1]);  // prefetch next tile
        const FR* bc = bfr[nt & 1];
        v8f acc = {};
        #pragma unroll
        for (int kt = 0; kt < 8; ++kt)
            acc = __builtin_amdgcn_wmma_f32_16x16x32_bf16(
                false, afr[kt].v, false, bc[kt].v, (short)0, acc, false, false);
        int n = nt * 16 + lm;
        float w2 = wn[n];
        #pragma unroll
        for (int v = 0; v < 8; ++v) {
            float dv = w2 - 2.0f * acc[v];
            if (dv < bestv[v]) { bestv[v] = dv; besti[v] = n; }
        }
    }
    // min+index reduce across the 16 lanes of each half (wave32), first-min ties
    #pragma unroll
    for (int off = 8; off >= 1; off >>= 1) {
        #pragma unroll
        for (int v = 0; v < 8; ++v) {
            float ov = __shfl_xor(bestv[v], off, 32);
            int   oi = __shfl_xor(besti[v], off, 32);
            if (ov < bestv[v] || (ov == bestv[v] && oi < besti[v])) {
                bestv[v] = ov; besti[v] = oi;
            }
        }
    }
    if (lm == 0) {
        #pragma unroll
        for (int v = 0; v < 8; ++v) idx_lds[wave * 16 + 8 * lh + v] = besti[v];
    }
    __syncthreads();

    // gather exact fp32 codewords -> emb output + bf16 copy for fc3
    for (int i = tid; i < 64 * 256; i += 128) {
        int lr = i >> 8, d = i & 255;
        int k = idx_lds[lr];
        float val = Wf[d * 256 + k];
        size_t r = rowBase + lr;
        size_t o = (r >> 3) * 2048 + (size_t)d * 8 + (r & 7);
        embOut[o] = val;
        Qbf[o] = f2bf(val);
    }
}

extern "C" void kernel_launch(void* const* d_in, const int* in_sizes, int n_in,
                              void* d_out, int out_size, void* d_ws, size_t ws_size,
                              hipStream_t stream) {
    (void)in_sizes; (void)n_in; (void)out_size; (void)ws_size;
    const float* x    = (const float*)d_in[0];
    const float* w1   = (const float*)d_in[1];
    const float* b1   = (const float*)d_in[2];
    const float* w2   = (const float*)d_in[3];
    const float* b2   = (const float*)d_in[4];
    const float* w3   = (const float*)d_in[5];
    const float* b3   = (const float*)d_in[6];
    const float* w4   = (const float*)d_in[7];
    const float* b4   = (const float*)d_in[8];
    const float* embw = (const float*)d_in[9];

    const int B = 32768, L = 784, H1 = 400, HID = 2048, K = 256;
    const int LP = 800, H1P = 416;                  // K dims padded to 32
    const int N1P = 448, N4P = 832;                 // W rows padded to 64-tiles

    char* ws = (char*)d_ws;
    size_t off = 0;
    auto alloc = [&](size_t bytes) -> char* {
        char* p = ws + off; off += (bytes + 255) & ~(size_t)255; return p;
    };
    // r0: x_bf (B x 800) then reused as q_bf (B x 2048); h_bf shared by h1/h3
    unsigned short* r0    = (unsigned short*)alloc((size_t)B * HID * 2);
    unsigned short* h_bf  = (unsigned short*)alloc((size_t)B * H1P * 2);
    unsigned short* ze_bf = (unsigned short*)alloc((size_t)B * HID * 2);
    unsigned short* w1b   = (unsigned short*)alloc((size_t)N1P * LP * 2);
    unsigned short* w2b   = (unsigned short*)alloc((size_t)HID * H1P * 2);
    unsigned short* w3b   = (unsigned short*)alloc((size_t)N1P * HID * 2);
    unsigned short* w4b   = (unsigned short*)alloc((size_t)N4P * H1P * 2);
    unsigned short* wtb   = (unsigned short*)alloc((size_t)K * K * 2);
    float*          wnp   = (float*)alloc((size_t)K * 4);

    float* out     = (float*)d_out;
    float* recon   = out;                          // B x 784
    float* ze_out  = out + (size_t)B * L;          // B x 2048
    float* emb_out = ze_out + (size_t)B * HID;     // B x 2048

    auto castPad = [&](const float* s, unsigned short* dp, int rows, int cols,
                       int colsPad, int rowsPad) {
        long long total = (long long)rowsPad * colsPad;
        cast_pad_kernel<<<(unsigned)((total + 255) / 256), 256, 0, stream>>>(
            s, dp, rows, cols, colsPad, total);
    };
    castPad(x,  r0,  B,   L,   LP,  B);     // x_bf   32768 x 800
    castPad(w1, w1b, H1,  L,   LP,  N1P);   // w1     448 x 800
    castPad(w2, w2b, HID, H1,  H1P, HID);   // w2     2048 x 416
    castPad(w3, w3b, H1,  HID, HID, N1P);   // w3     448 x 2048
    castPad(w4, w4b, L,   H1,  H1P, N4P);   // w4     832 x 416
    cast_transpose256<<<256, 256, 0, stream>>>(embw, wtb);
    wnorm_kernel<<<1, 256, 0, stream>>>(embw, wnp);

    dim3 blk(128);
    // fc1: h1 = relu(x @ w1^T + b1)
    gemm_kernel<1, false, true><<<dim3(B / 64, N1P / 64), blk, 0, stream>>>(
        r0, w1b, b1, nullptr, h_bf, H1, H1P, LP);
    // fc2: z_e = h1 @ w2^T + b2  (fp32 -> d_out, bf16 copy for quantize)
    gemm_kernel<0, true, true><<<dim3(B / 64, HID / 64), blk, 0, stream>>>(
        h_bf, w2b, b2, ze_out, ze_bf, HID, HID, H1P);
    // quantize: emb (fp32 -> d_out) + q bf16 (reuses r0)
    quantize_kernel<<<dim3((B * 8) / 64), blk, 0, stream>>>(
        ze_bf, wtb, embw, wnp, emb_out, r0);
    // fc3: h3 = relu(q @ w3^T + b3)
    gemm_kernel<1, false, true><<<dim3(B / 64, N1P / 64), blk, 0, stream>>>(
        r0, w3b, b3, nullptr, h_bf, H1, H1P, HID);
    // fc4: recon = sigmoid(h3 @ w4^T + b4)
    gemm_kernel<2, true, false><<<dim3(B / 64, N4P / 64), blk, 0, stream>>>(
        h_bf, w4b, b4, recon, nullptr, L, L, H1P);
}
